// AttnBlock_1924145348752
// MI455X (gfx1250) — matrix operand order repaired and checked by reading, not compiled, not run on hardware
//
#include <hip/hip_runtime.h>
#include <hip/hip_bf16.h>
#include <math.h>

#define B_ 4
#define C_ 512
#define T_ 4096
#define G_ 32
#define CPG_ 16
#define QSCALE 0.04419417382415922f   // 1/sqrt(512)

typedef _Float16 v16h __attribute__((ext_vector_type(16)));
typedef _Float16 v8h  __attribute__((ext_vector_type(8)));
typedef _Float16 v4h  __attribute__((ext_vector_type(4)));
typedef float    v8f  __attribute__((ext_vector_type(8)));
typedef int      v4i  __attribute__((vector_size(16)));   // matches builtin param

// ---- CDNA5 async global->LDS path (ASYNCcnt), with sync fallback ----
#if defined(__has_builtin)
#  if __has_builtin(__builtin_amdgcn_global_load_async_to_lds_b128) && \
      __has_builtin(__builtin_amdgcn_s_wait_asynccnt)
#    define USE_ASYNC_LDS 1
#  endif
#endif
#ifndef USE_ASYNC_LDS
#  define USE_ASYNC_LDS 0
#endif

#if USE_ASYNC_LDS
#pragma message("CDNA5 ASYNC LDS PATH: ON")
#else
#pragma message("CDNA5 ASYNC LDS PATH: OFF (sync fallback)")
#endif

// copy 16 bytes global -> LDS (per lane)
__device__ __forceinline__ void async_copy16(void* lds, const void* g) {
#if USE_ASYNC_LDS
  __builtin_amdgcn_global_load_async_to_lds_b128(
      (__attribute__((address_space(1))) v4i*)(g),
      (__attribute__((address_space(3))) v4i*)(lds), 0, 0);
#else
  *(v8h*)lds = *(const v8h*)g;
#endif
}
__device__ __forceinline__ void async_join() {
#if USE_ASYNC_LDS
  __builtin_amdgcn_s_wait_asynccnt(0);
#endif
}

// ---- WMMA fragment helpers (CDNA5 ISA 7.12.2 layouts, wave32) ----

// A matrix 16x32 f16 from [M][K] row-major storage (pitch ld).
__device__ __forceinline__ v16h load_a_frag(const _Float16* p, int ld) {
  const int lane = threadIdx.x & 31;
  const int m  = lane & 15;
  const int k0 = (lane < 16) ? 0 : 8;
  const _Float16* row = p + (size_t)m * ld + k0;
  v8h lo = *(const v8h*)(row);
  v8h hi = *(const v8h*)(row + 16);
  return __builtin_shufflevector(lo, hi, 0,1,2,3,4,5,6,7,8,9,10,11,12,13,14,15);
}

// B matrix 32x16 f16 from N-MAJOR storage Bt[N][K] (pitch ld): same k order as A (k0..k0+7, k0+16..k0+23).
__device__ __forceinline__ v16h load_b_frag_nm(const _Float16* p, int ld) {
  const int lane = threadIdx.x & 31;
  const int n  = lane & 15;
  const int k0 = (lane < 16) ? 0 : 8;
  const _Float16* col = p + (size_t)n * ld + k0;
  v8h lo = *(const v8h*)(col);
  v8h hi = *(const v8h*)(col + 16);
  return __builtin_shufflevector(lo, hi, 0,1,2,3,4,5,6,7,8,9,10,11,12,13,14,15);
}

__device__ __forceinline__ v8f wmma_f32_f16(v16h a, v16h b, v8f c) {
  return __builtin_amdgcn_wmma_f32_16x16x32_f16(false, a, false, b, (short)0, c,
                                                false, false);
}

// ---- Kernel 1: GroupNorm statistics (one block per (b,g)) ----
__global__ void gn_stats_kernel(const float* __restrict__ x,
                                float* __restrict__ mean,
                                float* __restrict__ rstd) {
  __shared__ float rs1[256], rs2[256];
  const int bg = blockIdx.x;                        // b*32+g spans contiguous mem
  const float4* base = (const float4*)(x + (size_t)bg * (CPG_ * T_));
  float s1 = 0.f, s2 = 0.f;
  for (int i = threadIdx.x; i < (CPG_ * T_) / 4; i += 256) {
    float4 v = base[i];
    s1 += v.x + v.y + v.z + v.w;
    s2 += v.x * v.x + v.y * v.y + v.z * v.z + v.w * v.w;
  }
  rs1[threadIdx.x] = s1; rs2[threadIdx.x] = s2;
  __syncthreads();
  for (int off = 128; off > 0; off >>= 1) {
    if (threadIdx.x < off) {
      rs1[threadIdx.x] += rs1[threadIdx.x + off];
      rs2[threadIdx.x] += rs2[threadIdx.x + off];
    }
    __syncthreads();
  }
  if (threadIdx.x == 0) {
    const float inv_n = 1.0f / (float)(CPG_ * T_);
    float m = rs1[0] * inv_n;
    float v = rs2[0] * inv_n - m * m;
    mean[bg] = m;
    rstd[bg] = rsqrtf(v + 1e-6f);
  }
}

// ---- Kernel 2: apply GN + cast to f16, h kept [C,T] ----
__global__ void gn_apply_kernel(const float* __restrict__ x,
                                const float* __restrict__ gw,
                                const float* __restrict__ gb,
                                const float* __restrict__ mean,
                                const float* __restrict__ rstd,
                                _Float16* __restrict__ h16) {
  size_t base = ((size_t)blockIdx.x * blockDim.x + threadIdx.x) * 4;
  if (base >= (size_t)B_ * C_ * T_) return;
  int c  = (int)((base / T_) % C_);
  int bg = (int)(base / ((size_t)CPG_ * T_));       // (b*32+g)
  float r = rstd[bg];
  float w = gw[c] * r;
  float bia = gb[c] - mean[bg] * w;                 // (x-m)*r*g+b = x*w+bia
  float4 v = *(const float4*)(x + base);
  v4h o;
  o[0] = (_Float16)(v.x * w + bia);
  o[1] = (_Float16)(v.y * w + bia);
  o[2] = (_Float16)(v.z * w + bia);
  o[3] = (_Float16)(v.w * w + bia);
  *(v4h*)(h16 + base) = o;
}

// ---- Kernel 3: tiled f16 transpose [B][C][T] -> [B][T][C] ----
__global__ void transpose_ct_tc_kernel(const _Float16* __restrict__ src,
                                       _Float16* __restrict__ dst) {
  __shared__ _Float16 tile[64][72];
  const int t0 = blockIdx.x * 64;
  const int c0 = blockIdx.y * 64;
  const int b  = blockIdx.z;
  const _Float16* s = src + (size_t)b * C_ * T_;
  _Float16* d = dst + (size_t)b * T_ * C_;
  const int r = threadIdx.x >> 2, seg = threadIdx.x & 3;
  async_copy16(&tile[r][seg * 16],
               s + (size_t)(c0 + r) * T_ + t0 + seg * 16);
  async_copy16(&tile[r][seg * 16 + 8],
               s + (size_t)(c0 + r) * T_ + t0 + seg * 16 + 8);
  async_join();
  __syncthreads();
  v8h o0, o1;
#pragma unroll
  for (int j = 0; j < 8; ++j) o0[j] = tile[seg * 16 + j][r];
#pragma unroll
  for (int j = 0; j < 8; ++j) o1[j] = tile[seg * 16 + 8 + j][r];
  *(v8h*)(d + (size_t)(t0 + r) * C_ + c0 + seg * 16)     = o0;
  *(v8h*)(d + (size_t)(t0 + r) * C_ + c0 + seg * 16 + 8) = o1;
}

// ---- Kernel 4: cast all four weight matrices to f16 ----
__global__ void cast_w_kernel(const float* __restrict__ wq, const float* __restrict__ wk,
                              const float* __restrict__ wv, const float* __restrict__ wo,
                              _Float16* __restrict__ w16) {
  int i = blockIdx.x * blockDim.x + threadIdx.x;
  if (i >= C_ * C_) return;
  w16[i]               = (_Float16)wq[i];
  w16[C_ * C_ + i]     = (_Float16)wk[i];
  w16[2 * C_ * C_ + i] = (_Float16)wv[i];
  w16[3 * C_ * C_ + i] = (_Float16)wo[i];
}

// ---- Kernel 5: fused QKV projection GEMM (WMMA, async double-buffered) ----
// Block tile [128 o x 64 t]; B tiles staged N-major [t][c] from h_tc.
// q -> [T,C] (1/sqrt(C) folded), k -> [T,C], v -> [C,T].
__global__ void qkv_gemm_kernel(const _Float16* __restrict__ h_tc,
                                const _Float16* __restrict__ w16,
                                const float* __restrict__ bq,
                                const float* __restrict__ bk,
                                const float* __restrict__ bv,
                                _Float16* __restrict__ qtc,
                                _Float16* __restrict__ ktc,
                                _Float16* __restrict__ vct) {
  __shared__ _Float16 Wt[2][128][40];   // A tiles [o][c], ping-pong
  __shared__ _Float16 Ht[2][64][40];    // B tiles N-major [t][c], ping-pong
  const int t0  = blockIdx.x * 64;
  const int o0  = blockIdx.y * 128;
  const int mat = blockIdx.z % 3;
  const int b   = blockIdx.z / 3;
  const int tid = threadIdx.x;
  const int w   = tid >> 5;
  const int lane = tid & 31;
  const int nn  = lane & 15;
  const int hi8 = (lane < 16) ? 0 : 8;
  const _Float16* wmat = w16 + (size_t)mat * C_ * C_;
  const _Float16* hb   = h_tc + (size_t)b * T_ * C_;

  const int wr = tid >> 1, whalf = tid & 1;   // W-tile staging map
  const int hr = tid >> 2, hseg  = tid & 3;   // H-tile staging map
  auto stage = [&](int bu, int kb) {
    const _Float16* wsrc = wmat + (size_t)(o0 + wr) * C_ + kb + whalf * 16;
    async_copy16(&Wt[bu][wr][whalf * 16],     wsrc);
    async_copy16(&Wt[bu][wr][whalf * 16 + 8], wsrc + 8);
    async_copy16(&Ht[bu][hr][hseg * 8],
                 hb + (size_t)(t0 + hr) * C_ + kb + hseg * 8);
  };

  v8f acc[4] = {};
  stage(0, 0);
  async_join();
  __syncthreads();
  int cur = 0;
  for (int kb = 0; kb < C_; kb += 32, cur ^= 1) {
    if (kb + 32 < C_) stage(cur ^ 1, kb + 32);   // overlap next DMA with WMMA
    v16h a = load_a_frag(&Wt[cur][w * 16][0], 40);
#pragma unroll
    for (int tt = 0; tt < 4; ++tt) {
      v16h bf = load_b_frag_nm(&Ht[cur][tt * 16][0], 40);
      acc[tt] = wmma_f32_f16(a, bf, acc[tt]);
    }
    async_join();
    __syncthreads();
  }

  const float* bias = (mat == 0) ? bq : (mat == 1) ? bk : bv;
#pragma unroll
  for (int tt = 0; tt < 4; ++tt) {
    int t = t0 + tt * 16 + nn;
    if (mat == 2) {                  // v -> [C,T]
#pragma unroll
      for (int r = 0; r < 8; ++r) {
        int o = o0 + w * 16 + r + hi8;
        vct[(size_t)b * C_ * T_ + (size_t)o * T_ + t] =
            (_Float16)(acc[tt][r] + bias[o]);
      }
    } else {                         // q,k -> [T,C] as one v8h per lane
      int ob = o0 + w * 16 + hi8;
      v8h o8;
#pragma unroll
      for (int r = 0; r < 8; ++r) {
        float val = acc[tt][r] + bias[ob + r];
        o8[r] = (_Float16)((mat == 0) ? val * QSCALE : val);
      }
      _Float16* outp = (mat == 0) ? qtc : ktc;
      *(v8h*)(outp + (size_t)b * T_ * C_ + (size_t)t * C_ + ob) = o8;
    }
  }
}

// ---- Kernel 6: flash attention (16 query rows/block, s-tiles of 32) ----
// One 40KB LDS buffer reused: K tile [32 s][520 c] for S, then V tile [512 c][40 s].
__global__ void flash_attn_kernel(const _Float16* __restrict__ qtc,
                                  const _Float16* __restrict__ ktc,
                                  const _Float16* __restrict__ vct,
                                  _Float16* __restrict__ h2ct) {
  extern __shared__ _Float16 dynsm[];
  _Float16* KsT = dynsm;             // [32][520]  N-major for S:  Kt[s][c]
  _Float16* VsT = dynsm;             // [512][40]  N-major for PV: Vt[c][s]
  __shared__ float    S_ls[16][32];
  __shared__ _Float16 P_ls[16][40];
  __shared__ float m_ls[16], l_ls[16], sc_ls[16];
  const int t0  = blockIdx.x * 16;
  const int b   = blockIdx.y;
  const int tid = threadIdx.x;
  const int w   = tid >> 5;
  const int lane = tid & 31;
  const int nn  = lane & 15;
  const int hi8 = (lane < 16) ? 0 : 8;

  const _Float16* qb = qtc + (size_t)b * T_ * C_ + (size_t)t0 * C_;
  const _Float16* kb = ktc + (size_t)b * T_ * C_;
  const _Float16* vb = vct + (size_t)b * C_ * T_;

  // Q fragments for this wave's 64-wide d-chunk, kept in registers
  v16h qa0 = load_a_frag(qb + w * 64,      C_);
  v16h qa1 = load_a_frag(qb + w * 64 + 32, C_);

  v8f acc[4] = {};
  if (tid < 16) { m_ls[tid] = -1e30f; l_ls[tid] = 0.f; }
  __syncthreads();

  const int kj = tid >> 3, kseg = tid & 7;  // K staging map (64 halfs/thread)
  const int vc = tid * 2;                   // V staging map (2 rows/thread)

  for (int s0 = 0; s0 < T_; s0 += 32) {
    { // stage K tile N-major via async DMA: KsT[j][c] = k_tc[s0+j][c]
      const _Float16* src = kb + (size_t)(s0 + kj) * C_ + kseg * 64;
      _Float16* dst = KsT + (size_t)kj * 520 + kseg * 64;
#pragma unroll
      for (int q8 = 0; q8 < 8; ++q8) async_copy16(dst + q8 * 8, src + q8 * 8);
    }
    { // zero S accumulator
      float* Sf = &S_ls[0][0];
      Sf[tid] = 0.f; Sf[tid + 256] = 0.f;
    }
    if (s0 + 32 < T_)  // pull next K tile toward cache (global_prefetch_b8)
      __builtin_prefetch(kb + (size_t)(s0 + 32 + kj) * C_ + kseg * 64, 0, 1);
    async_join();
    __syncthreads();

    // S partials: wave w covers d-chunk [w*64, w*64+64), both s-subtiles
#pragma unroll
    for (int ssub = 0; ssub < 2; ++ssub) {
      v8f sp = {};
      v16h b0 = load_b_frag_nm(KsT + (size_t)(ssub * 16) * 520 + w * 64,      520);
      sp = wmma_f32_f16(qa0, b0, sp);
      v16h b1 = load_b_frag_nm(KsT + (size_t)(ssub * 16) * 520 + w * 64 + 32, 520);
      sp = wmma_f32_f16(qa1, b1, sp);
#pragma unroll
      for (int r = 0; r < 8; ++r)
        atomicAdd(&S_ls[r + hi8][ssub * 16 + nn], sp[r]);  // ds_add_f32
    }
    __syncthreads();   // S done; K tile dead -> buffer reusable for V

    { // stage V tile via async DMA (overlaps with softmax below)
#pragma unroll
      for (int rr = 0; rr < 2; ++rr) {
        const _Float16* src = vb + (size_t)(vc + rr) * T_ + s0;
        _Float16* dst = VsT + (size_t)(vc + rr) * 40;
        async_copy16(dst,      src);
        async_copy16(dst + 8,  src + 8);
        async_copy16(dst + 16, src + 16);
        async_copy16(dst + 24, src + 24);
      }
    }
    // online softmax over the 32-wide tile (one thread per query row)
    if (tid < 16) {
      int row = tid;
      float mprev = m_ls[row];
      float mx = mprev;
#pragma unroll 8
      for (int j = 0; j < 32; ++j) mx = fmaxf(mx, S_ls[row][j]);
      float sc = __expf(mprev - mx);
      float sum = 0.f;
#pragma unroll 8
      for (int j = 0; j < 32; ++j) {
        float p = __expf(S_ls[row][j] - mx);
        P_ls[row][j] = (_Float16)p;
        sum += p;
      }
      l_ls[row]  = l_ls[row] * sc + sum;
      m_ls[row]  = mx;
      sc_ls[row] = sc;
    }
    async_join();
    __syncthreads();

    // rescale accumulators, then P @ V for this wave's 64 output channels
    v16h pa = load_a_frag(&P_ls[0][0], 40);
#pragma unroll
    for (int tt = 0; tt < 4; ++tt) {
#pragma unroll
      for (int r = 0; r < 8; ++r) acc[tt][r] *= sc_ls[r + hi8];
      v16h vf = load_b_frag_nm(VsT + (size_t)(w * 64 + tt * 16) * 40, 40);
      acc[tt] = wmma_f32_f16(pa, vf, acc[tt]);
    }
    __syncthreads();   // PV done before next iteration restages K
  }

  // finalize: divide by l; store h2 as [C,T] (v8h per lane, coalesced)
  float linv[8];
#pragma unroll
  for (int r = 0; r < 8; ++r) linv[r] = 1.0f / l_ls[r + hi8];
#pragma unroll
  for (int tt = 0; tt < 4; ++tt) {
    int c = w * 64 + tt * 16 + nn;
    v8h o8;
#pragma unroll
    for (int r = 0; r < 8; ++r) o8[r] = (_Float16)(acc[tt][r] * linv[r]);
    *(v8h*)(h2ct + (size_t)b * C_ * T_ + (size_t)c * T_ + t0 + hi8) = o8;
  }
}

// ---- Kernel 7: output projection + bias + residual (WMMA, async DB) ----
__global__ void out_proj_kernel(const _Float16* __restrict__ h2_tc,
                                const _Float16* __restrict__ wo16,
                                const float* __restrict__ bo,
                                const float* __restrict__ x,
                                float* __restrict__ out) {
  __shared__ _Float16 Wt[2][128][40];
  __shared__ _Float16 Ht[2][64][40];    // N-major [t][c]
  const int t0  = blockIdx.x * 64;
  const int o0  = blockIdx.y * 128;
  const int b   = blockIdx.z;
  const int tid = threadIdx.x;
  const int w   = tid >> 5;
  const int lane = tid & 31;
  const int nn  = lane & 15;
  const int hi8 = (lane < 16) ? 0 : 8;
  const _Float16* hb = h2_tc + (size_t)b * T_ * C_;

  const int wr = tid >> 1, whalf = tid & 1;
  const int hr = tid >> 2, hseg  = tid & 3;
  auto stage = [&](int bu, int kb) {
    const _Float16* wsrc = wo16 + (size_t)(o0 + wr) * C_ + kb + whalf * 16;
    async_copy16(&Wt[bu][wr][whalf * 16],     wsrc);
    async_copy16(&Wt[bu][wr][whalf * 16 + 8], wsrc + 8);
    async_copy16(&Ht[bu][hr][hseg * 8],
                 hb + (size_t)(t0 + hr) * C_ + kb + hseg * 8);
  };

  v8f acc[4] = {};
  stage(0, 0);
  async_join();
  __syncthreads();
  int cur = 0;
  for (int kb = 0; kb < C_; kb += 32, cur ^= 1) {
    if (kb + 32 < C_) stage(cur ^ 1, kb + 32);
    v16h a = load_a_frag(&Wt[cur][w * 16][0], 40);
#pragma unroll
    for (int tt = 0; tt < 4; ++tt) {
      v16h bf = load_b_frag_nm(&Ht[cur][tt * 16][0], 40);
      acc[tt] = wmma_f32_f16(a, bf, acc[tt]);
    }
    async_join();
    __syncthreads();
  }
#pragma unroll
  for (int tt = 0; tt < 4; ++tt) {
#pragma unroll
    for (int r = 0; r < 8; ++r) {
      int o = o0 + w * 16 + r + hi8;
      int t = t0 + tt * 16 + nn;
      size_t idx = (size_t)b * C_ * T_ + (size_t)o * T_ + t;
      out[idx] = x[idx] + bo[o] + acc[tt][r];
    }
  }
}

extern "C" void kernel_launch(void* const* d_in, const int* in_sizes, int n_in,
                              void* d_out, int out_size, void* d_ws, size_t ws_size,
                              hipStream_t stream) {
  (void)in_sizes; (void)n_in; (void)out_size; (void)ws_size;
  const float* x    = (const float*)d_in[0];
  const float* gn_w = (const float*)d_in[1];
  const float* gn_b = (const float*)d_in[2];
  const float* wq   = (const float*)d_in[3];
  const float* bq   = (const float*)d_in[4];
  const float* wk   = (const float*)d_in[5];
  const float* bk   = (const float*)d_in[6];
  const float* wv   = (const float*)d_in[7];
  const float* bv   = (const float*)d_in[8];
  const float* wo   = (const float*)d_in[9];
  const float* bo   = (const float*)d_in[10];
  float* out = (float*)d_out;

  char* ws = (char*)d_ws;
  float* mean = (float*)ws;
  float* rstd = mean + 128;
  size_t off = 1024;
  const size_t BCT2 = (size_t)B_ * C_ * T_ * sizeof(_Float16);
  _Float16* w16   = (_Float16*)(ws + off); off += (size_t)4 * C_ * C_ * sizeof(_Float16);
  _Float16* h_ct  = (_Float16*)(ws + off); off += BCT2;
  _Float16* h_tc  = (_Float16*)(ws + off); off += BCT2;
  _Float16* qtc   = (_Float16*)(ws + off); off += BCT2;
  _Float16* ktc   = (_Float16*)(ws + off); off += BCT2;
  _Float16* vct   = (_Float16*)(ws + off); off += BCT2;
  _Float16* h2_ct = (_Float16*)(ws + off); off += BCT2;
  _Float16* h2_tc = (_Float16*)(ws + off); off += BCT2;

  cast_w_kernel<<<dim3((C_ * C_ + 255) / 256), dim3(256), 0, stream>>>(
      wq, wk, wv, wo, w16);
  gn_stats_kernel<<<dim3(B_ * G_), dim3(256), 0, stream>>>(x, mean, rstd);
  gn_apply_kernel<<<dim3((B_ * C_ * T_ / 4 + 255) / 256), dim3(256), 0, stream>>>(
      x, gn_w, gn_b, mean, rstd, h_ct);
  transpose_ct_tc_kernel<<<dim3(T_ / 64, C_ / 64, B_), dim3(256), 0, stream>>>(
      h_ct, h_tc);
  qkv_gemm_kernel<<<dim3(T_ / 64, C_ / 128, 3 * B_), dim3(256), 0, stream>>>(
      h_tc, w16, bq, bk, bv, qtc, ktc, vct);
  flash_attn_kernel<<<dim3(T_ / 16, B_), dim3(256),
                      512 * 40 * sizeof(_Float16), stream>>>(
      qtc, ktc, vct, h2_ct);
  transpose_ct_tc_kernel<<<dim3(T_ / 64, C_ / 64, B_), dim3(256), 0, stream>>>(
      h2_ct, h2_tc);
  out_proj_kernel<<<dim3(T_ / 64, C_ / 128, B_), dim3(256), 0, stream>>>(
      h2_tc, w16 + (size_t)3 * C_ * C_, bo, x, out);
}
